// ActionEmbedding_89988154786527
// MI455X (gfx1250) — matrix operand order repaired and checked by reading, not compiled, not run on hardware
//
#include <hip/hip_runtime.h>

typedef _Float16 h16;
typedef __attribute__((ext_vector_type(8)))  _Float16 v8h;
typedef __attribute__((ext_vector_type(16))) _Float16 v16h;
typedef __attribute__((ext_vector_type(8)))  float    v8f;

#define TSTEPS      25
#define BSZ         16384
#define NDIN        13
#define NDOUT       9
#define INFER_ITERS 10
#define INFER_LR    0.1f
#define L2_LAMBDA   1e-3f
#define NWAVES      8

// Compiler-only ordering for same-wave LDS dependences (HW keeps LDS in-order per wave).
__device__ __forceinline__ void wsync() { __builtin_amdgcn_wave_barrier(); }

__device__ __forceinline__ v8f wmma_f16(v16h a, v16h b, v8f c) {
  return __builtin_amdgcn_wmma_f32_16x16x32_f16(
      false, a, false, b, (short)0, c, false, false);
}

// ---------------------------------------------------------------------------
// A-matrix (16x32 f16): lane m = lane&15; elements 0..7 -> K = kb + hi*8 + 0..7,
// elements 8..15 -> K = kb + 16 + hi*8 + 0..7  => two contiguous 16B chunks.
// kmaxA is a compile-time multiple of 16 => chunk validity folds at compile time.
// ---------------------------------------------------------------------------
__device__ __forceinline__ v16h loadA16(const h16* A, int stride, int kbase, int kmaxA) {
  int lane = threadIdx.x & 31;
  const h16* row = A + (lane & 15) * stride + ((lane & 16) ? 8 : 0);
  v8h z = {};
  v8h c0 = (kbase      < kmaxA) ? *(const v8h*)(row + kbase)      : z;
  v8h c1 = (kbase + 16 < kmaxA) ? *(const v8h*)(row + kbase + 16) : z;
  return __builtin_shufflevector(c0, c1, 0,1,2,3,4,5,6,7,8,9,10,11,12,13,14,15);
}

// ---------------------------------------------------------------------------
// B-matrix (32x16 f16): lane n = nb + lane&15; elements 0..7 -> K = kb + hi*8 + e, 8..15 -> K = kb + 16 + hi*8 + (e-8).
// With M stored so that B[k][n] = M[n][k], the 16 elements are contiguous:
// two ds_load_b128 per fetch. Rows are padded/zeroed so no masking is needed.
// ---------------------------------------------------------------------------
__device__ __forceinline__ v16h loadB_rows(const h16* M, int stride, int nbase, int kbase) {
  int lane = threadIdx.x & 31;
  const h16* p = M + (nbase + (lane & 15)) * stride + kbase + ((lane & 16) ? 8 : 0);
  v8h lo = *(const v8h*)(p);
  v8h hi = *(const v8h*)(p + 16);
  return __builtin_shufflevector(lo, hi, 0,1,2,3,4,5,6,7,8,9,10,11,12,13,14,15);
}

enum { M_NONE = 0, M_RELU = 1, M_RELUMASK = 2, M_TANHMASK = 3, M_TANH = 4 };

// D layout: lane = N col; VGPR r -> row M = r (lanes0-15) / r+8 (lanes16-31)
// HB: compile-time bias presence (no runtime pointer test -> unconditional load).
// For M_RELUMASK / M_TANHMASK, aux may alias D (read-then-write per element).
template <int MODE, bool HB>
__device__ __forceinline__ void storeD(h16* D, int dstride, int nbase, int noff,
                                       v8f acc, const float* bias,
                                       const h16* aux, int auxStride) {
  int lane = threadIdx.x & 31;
  int n  = lane & 15;
  int mb = (lane & 16) ? 8 : 0;
  int c  = nbase + n;
  float bv = HB ? bias[c] : 0.f;
#pragma unroll
  for (int r = 0; r < 8; ++r) {
    int m = mb + r;
    float v = acc[r] + bv;
    if (MODE == M_RELU)     v = fmaxf(v, 0.f);
    if (MODE == M_RELUMASK) v = ((float)aux[m * auxStride + c] > 0.f) ? v : 0.f;
    if (MODE == M_TANHMASK) { float h = (float)aux[m * auxStride + c]; v *= (1.f - h * h); }
    if (MODE == M_TANH)     v = tanhf(v);
    D[m * dstride + noff + c] = (h16)v;
  }
}

// Full 16xK @ KxN tile GEMM; A tiles hoisted (all loaded before any store).
template <int MODE, bool HB, int KT, int NT>
__device__ __forceinline__ void gemm_tile(const h16* A, int astride, int kmaxA,
                                          const h16* Bm, int bstride,
                                          h16* D, int dstride, int noff,
                                          const float* bias, const h16* aux, int auxStride) {
  v16h aT[KT];
#pragma unroll
  for (int kt = 0; kt < KT; ++kt) aT[kt] = loadA16(A, astride, kt * 32, kmaxA);
#pragma unroll
  for (int nt = 0; nt < NT; ++nt) {
    v8f acc = {};
#pragma unroll
    for (int kt = 0; kt < KT; ++kt)
      acc = wmma_f16(aT[kt], loadB_rows(Bm, bstride, nt * 16, kt * 32), acc);
    storeD<MODE, HB>(D, dstride, nt * 16, noff, acc, bias, aux, auxStride);
  }
}

// ---------------------------------------------------------------------------
// 256 threads = 8 waves per block (2 waves/SIMD); each wave owns 16 samples.
// Weights in LDS in both orientations; backward relu-masks applied in place
// so grads reuse sH1/sH2/sH3 ==> ~300 KB LDS total (< 320 KB WGP limit).
// ---------------------------------------------------------------------------
__global__ __launch_bounds__(32 * NWAVES)
void story_infer_kernel(const float* __restrict__ gx, const float* __restrict__ gy,
                        const float* gW1, const float* gb1,
                        const float* gW2, const float* gb2,
                        const float* gW3, const float* gb3,
                        const float* gW4, const float* gb4,
                        const float* gWih, const float* gbih,
                        const float* gWhh, const float* gbhh,
                        const float* gWd, const float* gbd,
                        float* __restrict__ partial) {
  // ---- shared weights (f16, dual orientation, padded rows) : ~142.5 KB ----
  __shared__ __align__(16) h16 sW2 [128 * 136];   // fwd:  B[k][n]=W2[n][k]
  __shared__ __align__(16) h16 sW2t[128 * 136];   // bwd:  B[k][n]=W2[k][n]
  __shared__ __align__(16) h16 sW3 [64 * 136];
  __shared__ __align__(16) h16 sW3t[128 * 72];
  __shared__ __align__(16) h16 sW4 [32 * 72];
  __shared__ __align__(16) h16 sW4t[64 * 40];
  __shared__ __align__(16) h16 sWih [32 * 72];    // K padded 45->64
  __shared__ __align__(16) h16 sWiht[64 * 40];
  __shared__ __align__(16) h16 sWhh [32 * 40];
  __shared__ __align__(16) h16 sW1f [128 * 40];   // K padded 4->32
  __shared__ __align__(16) h16 sW1t [16 * 136];
  __shared__ __align__(16) h16 sWd  [16 * 40];
  __shared__ __align__(16) h16 sWdt [32 * 40];
  __shared__ float sB1[128], sB2[128], sB3[64], sB4[32], sBih2[32], sBd[16];
  // ---- per-wave activation/grad state : ~19.5 KB x 8 ----
  __shared__ __align__(16) h16 shH1[NWAVES][16 * 136];
  __shared__ __align__(16) h16 shH2[NWAVES][16 * 136];
  __shared__ __align__(16) h16 shH3[NWAVES][16 * 72];
  __shared__ __align__(16) h16 shInp[NWAVES][16 * 72];   // [x(13)|w(32)|0-pad]
  __shared__ __align__(16) h16 shHid[NWAVES][16 * 40];
  __shared__ __align__(16) h16 shStoryH[NWAVES][16 * 40];
  __shared__ __align__(16) h16 shP[NWAVES][16 * 40];     // small grad ping
  __shared__ __align__(16) h16 shQ[NWAVES][16 * 40];     // small grad pong
  __shared__ float shStory[NWAVES][16 * 4];
  __shared__ float shY[NWAVES][16 * 20];

  const int tid  = threadIdx.x;
  const int lane = tid & 31;
  const int wv   = tid >> 5;
  const int sampBase = (blockIdx.x * NWAVES + wv) * 16;
  const float invB = 1.f / (float)BSZ;

  // ---- weight staging (once) ----
  for (int i = tid; i < 128 * 136; i += 32 * NWAVES) { int n = i / 136, k = i % 136;
    sW2 [i] = (k < 128) ? (h16)gW2[n * 128 + k] : (h16)0.f;
    sW2t[i] = (k < 128) ? (h16)gW2[k * 128 + n] : (h16)0.f; }
  for (int i = tid; i < 64 * 136; i += 32 * NWAVES) { int n = i / 136, k = i % 136;
    sW3 [i] = (k < 128) ? (h16)gW3[n * 128 + k] : (h16)0.f; }
  for (int i = tid; i < 128 * 72; i += 32 * NWAVES) { int a = i / 72, b = i % 72;
    sW3t[i] = (b < 64) ? (h16)gW3[b * 128 + a] : (h16)0.f; }
  for (int i = tid; i < 32 * 72; i += 32 * NWAVES) { int n = i / 72, k = i % 72;
    sW4 [i] = (k < 64) ? (h16)gW4[n * 64 + k] : (h16)0.f;
    sWih[i] = (k < 45) ? (h16)gWih[n * 45 + k] : (h16)0.f; }
  for (int i = tid; i < 64 * 40; i += 32 * NWAVES) { int a = i / 40, b = i % 40;
    sW4t [i] = (b < 32) ? (h16)gW4[b * 64 + a] : (h16)0.f;
    sWiht[i] = (a < 45 && b < 32) ? (h16)gWih[b * 45 + a] : (h16)0.f; }
  for (int i = tid; i < 32 * 40; i += 32 * NWAVES) { int n = i / 40, k = i % 40;
    sWhh[i] = (k < 32) ? (h16)gWhh[n * 32 + k] : (h16)0.f;
    sWdt[i] = (k < 9)  ? (h16)gWd[k * 32 + n]  : (h16)0.f; }
  for (int i = tid; i < 128 * 40; i += 32 * NWAVES) { int n = i / 40, k = i % 40;
    sW1f[i] = (k < 4) ? (h16)gW1[n * 4 + k] : (h16)0.f; }
  for (int i = tid; i < 16 * 136; i += 32 * NWAVES) { int a = i / 136, b = i % 136;
    sW1t[i] = (a < 4 && b < 128) ? (h16)gW1[b * 4 + a] : (h16)0.f; }
  for (int i = tid; i < 16 * 40; i += 32 * NWAVES) { int n = i / 40, k = i % 40;
    sWd[i] = (n < 9 && k < 32) ? (h16)gWd[n * 32 + k] : (h16)0.f; }
  for (int i = tid; i < 128; i += 32 * NWAVES) { sB1[i] = gb1[i]; sB2[i] = gb2[i]; }
  for (int i = tid; i < 64; i += 32 * NWAVES)  sB3[i] = gb3[i];
  for (int i = tid; i < 32; i += 32 * NWAVES)  { sB4[i] = gb4[i]; sBih2[i] = gbih[i] + gbhh[i]; }
  for (int i = tid; i < 16; i += 32 * NWAVES)  sBd[i] = (i < 9) ? gbd[i] : 0.f;

  // ---- per-wave pointers + init ----
  h16* sH1 = shH1[wv];  h16* sH2 = shH2[wv];  h16* sH3 = shH3[wv];
  h16* sInp = shInp[wv]; h16* sHid = shHid[wv]; h16* sStoryH = shStoryH[wv];
  h16* sP = shP[wv];    h16* sQ = shQ[wv];
  float* sStory = shStory[wv]; float* sY = shY[wv];

  for (int i = lane; i < 16 * 72; i += 32) sInp[i] = (h16)0.f;
  for (int i = lane; i < 16 * 40; i += 32) sStoryH[i] = (h16)0.f;
  for (int i = lane; i < 64; i += 32) sStory[i] = 0.f;
  __syncthreads();

  auto hyper_fwd = [&]() {
    gemm_tile<M_RELU, true, 1, 8>(sStoryH, 40, 32, sW1f, 40, sH1, 136, 0, sB1, nullptr, 0);
    wsync();
    gemm_tile<M_RELU, true, 4, 8>(sH1, 136, 128, sW2, 136, sH2, 136, 0, sB2, nullptr, 0);
    wsync();
    gemm_tile<M_RELU, true, 4, 4>(sH2, 136, 128, sW3, 136, sH3, 72, 0, sB3, nullptr, 0);
    wsync();
    gemm_tile<M_NONE, true, 2, 2>(sH3, 72, 64, sW4, 72, sInp, 72, 13, sB4, nullptr, 0);
    wsync();
  };

  auto rnn_cell = [&]() {
    v16h ia0 = loadA16(sInp, 72, 0, 64);
    v16h ia1 = loadA16(sInp, 72, 32, 64);
    v16h ha  = loadA16(sHid, 40, 0, 32);
    v8f a0 = {}, a1 = {};
    a0 = wmma_f16(ia0, loadB_rows(sWih, 72, 0, 0),  a0);
    a0 = wmma_f16(ia1, loadB_rows(sWih, 72, 0, 32), a0);
    a0 = wmma_f16(ha,  loadB_rows(sWhh, 40, 0, 0),  a0);
    a1 = wmma_f16(ia0, loadB_rows(sWih, 72, 16, 0),  a1);
    a1 = wmma_f16(ia1, loadB_rows(sWih, 72, 16, 32), a1);
    a1 = wmma_f16(ha,  loadB_rows(sWhh, 40, 16, 0),  a1);
    wsync();
    storeD<M_TANH, true>(sHid, 40, 0,  0, a0, sBih2, nullptr, 0);
    storeD<M_TANH, true>(sHid, 40, 16, 0, a1, sBih2, nullptr, 0);
    wsync();
  };

  // ====================== Phase 1: 25 x 10 inference steps ==================
#pragma unroll 1
  for (int t = 0; t < TSTEPS; ++t) {
    for (int i = lane; i < 16 * NDIN; i += 32) { int m = i / NDIN, j = i % NDIN;
      sInp[m * 72 + j] = (h16)gx[((size_t)t * BSZ + sampBase + m) * NDIN + j]; }
    for (int i = lane; i < 16 * 16; i += 32) { int m = i >> 4, j = i & 15;
      sY[m * 20 + j] = (j < 9) ? gy[((size_t)t * BSZ + sampBase + m) * NDOUT + j] : 0.f; }
    for (int i = lane; i < 16 * 40; i += 32) sHid[i] = (h16)0.f;
    wsync();

#pragma unroll 1
    for (int it = 0; it < INFER_ITERS; ++it) {
      hyper_fwd();
      rnn_cell();   // sHid := h_new

      {  // pred + UNSCALED grad: g = 2*(pred - y) -> sP (16x16 tile, stride 40)
        v16h pa = loadA16(sHid, 40, 0, 32);
        v8f acc = {};
        acc = wmma_f16(pa, loadB_rows(sWd, 40, 0, 0), acc);
        int n = lane & 15, mb = (lane & 16) ? 8 : 0;
#pragma unroll
        for (int r = 0; r < 8; ++r) {
          int m = mb + r;
          float p  = acc[r] + sBd[n];
          float yv = sY[m * 20 + n];
          float g  = (n < NDOUT) ? 2.f * (p - yv) : 0.f;
          sP[m * 40 + n] = (h16)g;
        }
      }
      wsync();

      // da  = (dpred @ Wd) * (1 - h_new^2)            sP -> sQ
      gemm_tile<M_TANHMASK, false, 1, 2>(sP, 40, 16, sWdt, 40, sQ, 40, 0, nullptr, sHid, 40);
      wsync();
      // dw  = da @ Wih[:, 13:45]                      sQ -> sP
      gemm_tile<M_NONE, false, 1, 2>(sQ, 40, 32, sWiht + 13 * 40, 40, sP, 40, 0, nullptr, nullptr, 0);
      wsync();
      // dh3 = (dw @ W4) * (h3 > 0)                    sP -> sH3 (mask in place)
      gemm_tile<M_RELUMASK, false, 1, 4>(sP, 40, 32, sW4t, 40, sH3, 72, 0, nullptr, sH3, 72);
      wsync();
      // dh2 = (dh3 @ W3) * (h2 > 0)                   sH3 -> sH2 (mask in place)
      gemm_tile<M_RELUMASK, false, 2, 8>(sH3, 72, 64, sW3t, 72, sH2, 136, 0, nullptr, sH2, 136);
      wsync();
      // dh1 = (dh2 @ W2) * (h1 > 0)                   sH2 -> sH1 (mask in place)
      gemm_tile<M_RELUMASK, false, 4, 8>(sH2, 136, 128, sW2t, 136, sH1, 136, 0, nullptr, sH1, 136);
      wsync();

      {  // dstory = dh1 @ W1 ; SGD step in f32 with 1/B + L2 here
        v16h aT[4];
#pragma unroll
        for (int kt = 0; kt < 4; ++kt) aT[kt] = loadA16(sH1, 136, kt * 32, 128);
        v8f acc = {};
#pragma unroll
        for (int kt = 0; kt < 4; ++kt)
          acc = wmma_f16(aT[kt], loadB_rows(sW1t, 136, 0, kt * 32), acc);
        int n = lane & 15, mb = (lane & 16) ? 8 : 0;
        if (n < 4) {
#pragma unroll
          for (int r = 0; r < 8; ++r) {
            int m = mb + r;
            float s = sStory[m * 4 + n];
            float g = acc[r] * invB + 2.f * L2_LAMBDA * invB * s;
            float ns = s - INFER_LR * g;
            sStory[m * 4 + n] = ns;
            sStoryH[m * 40 + n] = (h16)ns;
          }
        }
      }
      wsync();
    }
  }

  // ====================== Phase 2: final rollout ============================
  hyper_fwd();   // w = hyper(final story) -> sInp cols 13..44
  for (int i = lane; i < 16 * 40; i += 32) sHid[i] = (h16)0.f;
  wsync();

  float errAcc = 0.f;
#pragma unroll 1
  for (int t = 0; t < TSTEPS; ++t) {
    for (int i = lane; i < 16 * NDIN; i += 32) { int m = i / NDIN, j = i % NDIN;
      sInp[m * 72 + j] = (h16)gx[((size_t)t * BSZ + sampBase + m) * NDIN + j]; }
    for (int i = lane; i < 16 * 16; i += 32) { int m = i >> 4, j = i & 15;
      sY[m * 20 + j] = (j < 9) ? gy[((size_t)t * BSZ + sampBase + m) * NDOUT + j] : 0.f; }
    wsync();
    rnn_cell();
    {
      v16h pa = loadA16(sHid, 40, 0, 32);
      v8f acc = {};
      acc = wmma_f16(pa, loadB_rows(sWd, 40, 0, 0), acc);
      int n = lane & 15, mb = (lane & 16) ? 8 : 0;
#pragma unroll
      for (int r = 0; r < 8; ++r) {
        int m = mb + r;
        float p = acc[r] + sBd[n];
        float d = p - sY[m * 20 + n];
        errAcc += (n < NDOUT) ? d * d : 0.f;
      }
    }
    wsync();
  }

  // wave-wide reduction -> per-wave partial (deterministic)
#pragma unroll
  for (int off = 16; off > 0; off >>= 1)
    errAcc += __shfl_xor(errAcc, off, 32);
  if (lane == 0) partial[blockIdx.x * NWAVES + wv] = errAcc;
}

__global__ __launch_bounds__(256)
void reduce_kernel(const float* __restrict__ partial, float* __restrict__ out, int n) {
  __shared__ float buf[256];
  float s = 0.f;
  for (int i = threadIdx.x; i < n; i += 256) s += partial[i];
  buf[threadIdx.x] = s;
  __syncthreads();
  for (int off = 128; off > 0; off >>= 1) {
    if (threadIdx.x < off) buf[threadIdx.x] += buf[threadIdx.x + off];
    __syncthreads();
  }
  if (threadIdx.x == 0) out[0] = buf[0] * (1.f / (float)BSZ);
}

extern "C" void kernel_launch(void* const* d_in, const int* in_sizes, int n_in,
                              void* d_out, int out_size, void* d_ws, size_t ws_size,
                              hipStream_t stream) {
  const float* gx   = (const float*)d_in[0];
  const float* gy   = (const float*)d_in[1];
  const float* gW1  = (const float*)d_in[2];
  const float* gb1  = (const float*)d_in[3];
  const float* gW2  = (const float*)d_in[4];
  const float* gb2  = (const float*)d_in[5];
  const float* gW3  = (const float*)d_in[6];
  const float* gb3  = (const float*)d_in[7];
  const float* gW4  = (const float*)d_in[8];
  const float* gb4  = (const float*)d_in[9];
  const float* gWih = (const float*)d_in[10];
  const float* gbih = (const float*)d_in[11];
  const float* gWhh = (const float*)d_in[12];
  const float* gbhh = (const float*)d_in[13];
  const float* gWd  = (const float*)d_in[14];
  const float* gbd  = (const float*)d_in[15];

  float* partial = (float*)d_ws;
  const int nBlocks = BSZ / (16 * NWAVES);   // 128 blocks x 8 waves x 16 samples

  story_infer_kernel<<<dim3(nBlocks), dim3(32 * NWAVES), 0, stream>>>(
      gx, gy, gW1, gb1, gW2, gb2, gW3, gb3, gW4, gb4,
      gWih, gbih, gWhh, gbhh, gWd, gbd, partial);
  reduce_kernel<<<dim3(1), dim3(256), 0, stream>>>(partial, (float*)d_out,
                                                   nBlocks * NWAVES);
}
